// GCN_Net_53901839565316
// MI455X (gfx1250) — compile-verified, matrix operand order checked by reading, not run on hardware
//
#include <hip/hip_runtime.h>
#include <hip/hip_bf16.h>
#include <math.h>

typedef __attribute__((ext_vector_type(16))) __bf16 v16bf;
typedef __attribute__((ext_vector_type(8)))  float  v8f;

#define N_NODES 100000
#define N_EDGES 1600000
#define F_IN    256
#define F_HID   128
#define F_CLS   40
#define F_CLSP  48   // padded class dim for WMMA (multiple of 16)

static __device__ __forceinline__ void atomic_add_f32(float* p, float v) {
  __hip_atomic_fetch_add(p, v, __ATOMIC_RELAXED, __HIP_MEMORY_SCOPE_AGENT);
}

// ---------- degree / normalization ----------
__global__ void deg_init_kernel(float* deg, int n) {
  int i = blockIdx.x * blockDim.x + threadIdx.x;
  if (i < n) deg[i] = 1.0f;  // self loop
}

__global__ void deg_count_kernel(const int* __restrict__ dst, float* deg, int e) {
  int i = blockIdx.x * blockDim.x + threadIdx.x;
  if (i < e) atomic_add_f32(&deg[dst[i]], 1.0f);
}

__global__ void deg_rsqrt_kernel(float* deg, int n) {
  int i = blockIdx.x * blockDim.x + threadIdx.x;
  if (i < n) deg[i] = rsqrtf(deg[i]);  // deg >= 1 always
}

// ---------- conversions ----------
__global__ void cvt_bf16_kernel(const float* __restrict__ in, __bf16* __restrict__ out, size_t n) {
  size_t i = (size_t)blockIdx.x * blockDim.x + threadIdx.x;
  if (i < n) out[i] = (__bf16)in[i];
}

// Wt[n*K + k] = (n < Nsrc) ? W[k*Nsrc + n] : 0   over Npad*K elements
__global__ void transpose_bf16_kernel(const float* __restrict__ W, __bf16* __restrict__ Wt,
                                      int K, int Nsrc, int Npad) {
  int idx = blockIdx.x * blockDim.x + threadIdx.x;
  if (idx >= Npad * K) return;
  int n = idx / K, k = idx - n * K;
  Wt[(size_t)n * K + k] = (n < Nsrc) ? (__bf16)W[(size_t)k * Nsrc + n] : (__bf16)0.0f;
}

// ---------- WMMA GEMM: C[M x ldc] = A[M x K](bf16,row) * Bt[N x K](bf16,row)^T ----------
__global__ __launch_bounds__(256)
void gemm_bf16_wmma_kernel(const __bf16* __restrict__ A, const __bf16* __restrict__ Bt,
                           float* __restrict__ C, int K, int nTilesM, int nTilesN, int ldc) {
  int wid  = (blockIdx.x * blockDim.x + threadIdx.x) >> 5;
  int nTiles = nTilesM * nTilesN;
  if (wid >= nTiles) return;
  int tm = wid / nTilesN;
  int tn = wid - tm * nTilesN;
  int lane  = threadIdx.x & 31;
  int kgrp  = lane >> 4;     // 0: K-halves 0..7, 1: K-halves 8..15 per 16-group
  int idx16 = lane & 15;     // A: row-in-tile, B: col-in-tile

  const __bf16* arow = A  + (size_t)(tm * 16 + idx16) * K + kgrp * 8;
  const __bf16* brow = Bt + (size_t)(tn * 16 + idx16) * K + kgrp * 8;

  v8f acc = {};
  for (int k0 = 0; k0 < K; k0 += 32) {
    union { v16bf v; uint4 u[2]; } a, b;
    a.u[0] = *(const uint4*)(arow + k0);        // K = k0 + kgrp*8 + 0..7
    a.u[1] = *(const uint4*)(arow + k0 + 16);   // K = k0 + 16 + kgrp*8 + 0..7
    b.u[0] = *(const uint4*)(brow + k0);
    b.u[1] = *(const uint4*)(brow + k0 + 16);
    if (k0 + 32 < K) {
      __builtin_prefetch(arow + k0 + 32, 0, 3);
      __builtin_prefetch(brow + k0 + 32, 0, 3);
    }
    acc = __builtin_amdgcn_wmma_f32_16x16x32_bf16(false, a.v, false, b.v,
                                                  (short)0, acc, false, false);
  }

  // C/D layout: VGPR r -> row r + 8*kgrp (lanes 16-31 hold M=8..15), col = lane%16
  float* crow = C + (size_t)(tm * 16 + kgrp * 8) * ldc + tn * 16 + idx16;
  #pragma unroll
  for (int r = 0; r < 8; ++r) crow[(size_t)r * ldc] = acc[r];
}

// ---------- aggregation ----------
// out[i*ldo + f] = dinv[i]^2 * h[i*ldh + f]   (self-loop term), over nNodes*Fout
__global__ void agg_init_kernel(const float* __restrict__ dinv, const float* __restrict__ h,
                                float* __restrict__ out, int Fout, int ldh, int ldo, int nNodes) {
  long long idx = (long long)blockIdx.x * blockDim.x + threadIdx.x;
  long long total = (long long)nNodes * Fout;
  if (idx >= total) return;
  int i = (int)(idx / Fout);
  int f = (int)(idx - (long long)i * Fout);
  float d = dinv[i];
  out[(size_t)i * ldo + f] = d * d * h[(size_t)i * ldh + f];
}

// one wave per edge: out[dst] += dinv[s]*dinv[d] * h[src]
__global__ void scatter_kernel(const int* __restrict__ src, const int* __restrict__ dst,
                               const float* __restrict__ dinv, const float* __restrict__ h,
                               float* __restrict__ out, int e, int Fout, int ldh, int ldo) {
  int wave = (int)(((long long)blockIdx.x * blockDim.x + threadIdx.x) >> 5);
  int lane = threadIdx.x & 31;
  if (wave >= e) return;
  int s = src[wave], d = dst[wave];
  float w = dinv[s] * dinv[d];
  const float* hs = h + (size_t)s * ldh;
  float* od = out + (size_t)d * ldo;
  for (int f = lane; f < Fout; f += 32)
    atomic_add_f32(&od[f], w * hs[f]);
}

// h1b = bf16(relu(out1 + b1)), over N*128
__global__ void bias_relu_cvt_kernel(const float* __restrict__ out1, const float* __restrict__ b1,
                                     __bf16* __restrict__ h1b, int nNodes) {
  long long idx = (long long)blockIdx.x * blockDim.x + threadIdx.x;
  long long total = (long long)nNodes * F_HID;
  if (idx >= total) return;
  int f = (int)(idx & (F_HID - 1));
  float v = out1[idx] + b1[f];
  h1b[idx] = (__bf16)fmaxf(v, 0.0f);
}

// d_out = log_softmax(out2 + b2), one thread per node
__global__ void logsoftmax_kernel(const float* __restrict__ out2, const float* __restrict__ b2,
                                  float* __restrict__ y, int nNodes) {
  int i = blockIdx.x * blockDim.x + threadIdx.x;
  if (i >= nNodes) return;
  const float* z = out2 + (size_t)i * F_CLS;
  float v[F_CLS];
  float m = -INFINITY;
  #pragma unroll
  for (int f = 0; f < F_CLS; ++f) { v[f] = z[f] + b2[f]; m = fmaxf(m, v[f]); }
  float s = 0.0f;
  #pragma unroll
  for (int f = 0; f < F_CLS; ++f) s += expf(v[f] - m);
  float lse = m + logf(s);
  float* yo = y + (size_t)i * F_CLS;
  #pragma unroll
  for (int f = 0; f < F_CLS; ++f) yo[f] = v[f] - lse;
}

extern "C" void kernel_launch(void* const* d_in, const int* in_sizes, int n_in,
                              void* d_out, int out_size, void* d_ws, size_t ws_size,
                              hipStream_t stream) {
  const float* x   = (const float*)d_in[0];
  const int*   ei  = (const int*)d_in[1];
  const float* W1  = (const float*)d_in[2];
  const float* b1  = (const float*)d_in[3];
  const float* W2  = (const float*)d_in[4];
  const float* b2  = (const float*)d_in[5];
  float* y = (float*)d_out;
  const int* srcI = ei;            // edge_index[0]
  const int* dstI = ei + N_EDGES;  // edge_index[1]

  // workspace layout (aliased regions, all 256B-aligned)
  char* ws = (char*)d_ws;
  size_t o = 0;
  auto alloc = [&](size_t bytes) -> char* {
    o = (o + 255) & ~(size_t)255;
    char* p = ws + o; o += bytes; return p;
  };
  float* dinv = (float*)alloc((size_t)N_NODES * 4);
  char* R1 = alloc((size_t)N_NODES * F_IN * 2);   // xb(bf16) -> out1(f32) -> h2(f32, ld 48)
  char* R2 = alloc((size_t)N_NODES * F_HID * 4);  // h1(f32) -> h1b(bf16) + out2(f32)
  __bf16* W1t = (__bf16*)alloc((size_t)F_HID * F_IN * 2);
  __bf16* W2t = (__bf16*)alloc((size_t)F_CLSP * F_HID * 2);

  __bf16* xb   = (__bf16*)R1;
  float*  out1 = (float*)R1;
  float*  h2   = (float*)R1;
  float*  h1   = (float*)R2;
  __bf16* h1b  = (__bf16*)R2;
  float*  out2 = (float*)(R2 + (size_t)N_NODES * F_HID * 2);

  const int B = 256;

  // 1) degree -> dinv
  deg_init_kernel<<<(N_NODES + B - 1) / B, B, 0, stream>>>(dinv, N_NODES);
  deg_count_kernel<<<(N_EDGES + B - 1) / B, B, 0, stream>>>(dstI, dinv, N_EDGES);
  deg_rsqrt_kernel<<<(N_NODES + B - 1) / B, B, 0, stream>>>(dinv, N_NODES);

  // 2) weights -> transposed bf16
  transpose_bf16_kernel<<<(F_HID * F_IN + B - 1) / B, B, 0, stream>>>(W1, W1t, F_IN, F_HID, F_HID);
  transpose_bf16_kernel<<<(F_CLSP * F_HID + B - 1) / B, B, 0, stream>>>(W2, W2t, F_HID, F_CLS, F_CLSP);

  // 3) x -> bf16
  size_t nx = (size_t)N_NODES * F_IN;
  cvt_bf16_kernel<<<(int)((nx + B - 1) / B), B, 0, stream>>>(x, xb, nx);

  // 4) GEMM1: h1[N x 128] = xb[N x 256] @ W1   (8 WMMAs per wave)
  {
    int tiles = (N_NODES / 16) * (F_HID / 16);       // 6250 * 8
    int blocks = (tiles + 7) / 8;                     // 8 waves / block
    gemm_bf16_wmma_kernel<<<blocks, B, 0, stream>>>(xb, W1t, h1, F_IN, N_NODES / 16, F_HID / 16, F_HID);
  }

  // 5) aggregate layer 1: out1 = D^-1/2 (A+I) D^-1/2 h1
  {
    long long total = (long long)N_NODES * F_HID;
    agg_init_kernel<<<(int)((total + B - 1) / B), B, 0, stream>>>(dinv, h1, out1, F_HID, F_HID, F_HID, N_NODES);
    int blocks = (N_EDGES + 7) / 8;  // one wave per edge
    scatter_kernel<<<blocks, B, 0, stream>>>(srcI, dstI, dinv, h1, out1, N_EDGES, F_HID, F_HID, F_HID);
    bias_relu_cvt_kernel<<<(int)((total + B - 1) / B), B, 0, stream>>>(out1, b1, h1b, N_NODES);
  }

  // 6) GEMM2: h2[N x 48] = h1b[N x 128] @ W2pad   (4 WMMAs per wave)
  {
    int tiles = (N_NODES / 16) * (F_CLSP / 16);      // 6250 * 3
    int blocks = (tiles + 7) / 8;
    gemm_bf16_wmma_kernel<<<blocks, B, 0, stream>>>(h1b, W2t, h2, F_HID, N_NODES / 16, F_CLSP / 16, F_CLSP);
  }

  // 7) aggregate layer 2: out2 (stride 40) from h2 (stride 48)
  {
    long long total = (long long)N_NODES * F_CLS;
    agg_init_kernel<<<(int)((total + B - 1) / B), B, 0, stream>>>(dinv, h2, out2, F_CLS, F_CLSP, F_CLS, N_NODES);
    int blocks = (N_EDGES + 7) / 8;
    scatter_kernel<<<blocks, B, 0, stream>>>(srcI, dstI, dinv, h2, out2, N_EDGES, F_CLS, F_CLSP, F_CLS);
  }

  // 8) bias + log_softmax -> d_out
  logsoftmax_kernel<<<(N_NODES + B - 1) / B, B, 0, stream>>>(out2, b2, y, N_NODES);
}
